// GameProcessor_67980742361453
// MI455X (gfx1250) — matrix operand order repaired and checked by reading, not compiled, not run on hardware
//
#include <hip/hip_runtime.h>
#include <math.h>

typedef __attribute__((ext_vector_type(16))) __bf16 v16bf;
typedef __attribute__((ext_vector_type(8)))  float  v8f;
typedef __attribute__((ext_vector_type(4)))  unsigned int v4u;
typedef __attribute__((ext_vector_type(8)))  int    v8i;
typedef __attribute__((ext_vector_type(4)))  int    v4i;

#define B_ROWS 16384
#define DDIM   512
#define FDIM   28
#define KPAD   1056      /* 1053 padded to multiple of 32 */
#define G3     1536
#define LN_EPS 1e-5f

/* ---------- helpers ---------- */

__device__ __forceinline__ unsigned short f2bf(float f) {
    unsigned int u = __float_as_uint(f);
    unsigned int r = u + 0x7FFFu + ((u >> 16) & 1u);   /* round-nearest-even */
    return (unsigned short)(r >> 16);
}

__device__ __forceinline__ float sigmoidf(float x) {
    return 1.0f / (1.0f + expf(-x));
}

/* A fragment: 16x32 bf16, row-major source.
   Lanes 0-15: row m0+lane, K=k0..k0+15 contiguous; lanes 16-31: row m0+lane-16,
   K=k0+16..k0+31 (matches B fragment order). */
__device__ __forceinline__ v16bf load_A(const unsigned short* __restrict__ base,
                                        int ld, int m0, int k0, int lane) {
    int row = m0 + (lane & 15);
    int kb  = k0 + ((lane & 16) ? 16 : 0);
    const unsigned short* p = base + row * ld + kb;
    union { v16bf v; uint4 u[2]; } t;
    t.u[0] = *(const uint4*)(p);
    t.u[1] = *(const uint4*)(p + 8);
    return t.v;
}

/* B fragment: 32x16 bf16 from row-major weight [N,K] (ld = K).
   Lanes 0-15: N=n0+lane, K=k0..k0+15 contiguous; lanes 16-31: K=k0+16..k0+31. */
__device__ __forceinline__ v16bf load_B(const unsigned short* __restrict__ w,
                                        int ld, int n0, int k0, int lane) {
    int n  = n0 + (lane & 15);
    int kb = k0 + ((lane & 16) ? 16 : 0);
    const unsigned short* p = w + n * ld + kb;
    union { v16bf v; uint4 u[2]; } t;
    t.u[0] = *(const uint4*)(p);
    t.u[1] = *(const uint4*)(p + 8);
    return t.v;
}

/* B fragment from an LDS-staged tile (row-major [rows x DDIM]); emits ds_load_b128 */
__device__ __forceinline__ v16bf load_B_sh(const unsigned short* sh,
                                           int rowbase, int k0, int lane) {
    int r  = rowbase + (lane & 15);
    int kb = k0 + ((lane & 16) ? 16 : 0);
    const unsigned short* p = sh + r * DDIM + kb;
    union { v16bf v; uint4 u[2]; } t;
    t.u[0] = *(const uint4*)(p);
    t.u[1] = *(const uint4*)(p + 8);
    return t.v;
}

#define WMMA_BF16(A, Bm, C) \
    __builtin_amdgcn_wmma_f32_16x16x32_bf16(false, (A), false, (Bm), (short)0, (C), false, false)

/* ---------- packing / conversion kernels ---------- */

__global__ void pack_x_kernel(const float* __restrict__ te, const float* __restrict__ oe,
                              const float* __restrict__ gf, const float* __restrict__ won,
                              unsigned short* __restrict__ Xb, int total) {
    int idx = blockIdx.x * blockDim.x + threadIdx.x;
    if (idx >= total) return;
    int b = idx / KPAD;
    int k = idx - b * KPAD;
    float v;
    if (k < DDIM)                  v = te[b * DDIM + k];
    else if (k < 2 * DDIM)         v = oe[b * DDIM + (k - DDIM)];
    else if (k < 2 * DDIM + FDIM)  v = gf[b * FDIM + (k - 2 * DDIM)];
    else if (k == 2 * DDIM + FDIM) v = won[b];
    else                           v = 0.0f;
    Xb[idx] = f2bf(v);
}

__global__ void convert_pad_kernel(const float* __restrict__ src, unsigned short* __restrict__ dst,
                                   int srcK, int dstK, int total) {
    int idx = blockIdx.x * blockDim.x + threadIdx.x;
    if (idx >= total) return;
    int r = idx / dstK;
    int k = idx - r * dstK;
    dst[idx] = (k < srcK) ? f2bf(src[r * srcK + k]) : (unsigned short)0;
}

/* ---------- GEMM 1: projected = relu(X @ Wp^T + b) ---------- */
/* grid (B/128, DDIM/64), block 128 (4 waves); wave = 32 rows x 64 cols, 8 accum */
__global__ __launch_bounds__(128)
void gemm_proj_kernel(const unsigned short* __restrict__ Xb,   /* [B, KPAD]    */
                      const unsigned short* __restrict__ Wp,   /* [DDIM, KPAD] */
                      const float* __restrict__ bias,
                      float* __restrict__ gc_out,              /* [B, DDIM] fp32 */
                      unsigned short* __restrict__ proj_bf) {  /* [B, DDIM] bf16 */
    int lane = threadIdx.x & 31;
    int wave = threadIdx.x >> 5;
    int m0 = (blockIdx.x * 8 + wave * 2) * 16;   /* rows m0 .. m0+31 */
    int n0 = blockIdx.y * 64;

    v8f acc[2][4] = {};
    for (int k0 = 0; k0 < KPAD; k0 += 32) {
        if (k0 + 32 < KPAD) {   /* CDNA5 prefetch of next K tile (global_prefetch_b8) */
            __builtin_prefetch(Xb + (m0 + (lane & 15)) * KPAD + k0 + 32, 0, 0);
            __builtin_prefetch(Wp + (n0 + (lane & 15)) * KPAD + k0 + 32, 0, 0);
        }
        v16bf a0 = load_A(Xb, KPAD, m0,      k0, lane);
        v16bf a1 = load_A(Xb, KPAD, m0 + 16, k0, lane);
        v16bf b0 = load_B(Wp, KPAD, n0,      k0, lane);
        v16bf b1 = load_B(Wp, KPAD, n0 + 16, k0, lane);
        v16bf b2 = load_B(Wp, KPAD, n0 + 32, k0, lane);
        v16bf b3 = load_B(Wp, KPAD, n0 + 48, k0, lane);
        acc[0][0] = WMMA_BF16(a0, b0, acc[0][0]);
        acc[0][1] = WMMA_BF16(a0, b1, acc[0][1]);
        acc[0][2] = WMMA_BF16(a0, b2, acc[0][2]);
        acc[0][3] = WMMA_BF16(a0, b3, acc[0][3]);
        acc[1][0] = WMMA_BF16(a1, b0, acc[1][0]);
        acc[1][1] = WMMA_BF16(a1, b1, acc[1][1]);
        acc[1][2] = WMMA_BF16(a1, b2, acc[1][2]);
        acc[1][3] = WMMA_BF16(a1, b3, acc[1][3]);
    }
    int col = lane & 15;
#pragma unroll
    for (int mi = 0; mi < 2; mi++) {
        int rbase = m0 + mi * 16 + ((lane & 16) ? 8 : 0);
#pragma unroll
        for (int t = 0; t < 4; t++) {
            int n = n0 + t * 16 + col;
            float bv = bias[n];
#pragma unroll
            for (int r = 0; r < 8; r++) {
                float v = acc[mi][t][r] + bv;
                v = v > 0.0f ? v : 0.0f;
                int row = rbase + r;
                gc_out[row * DDIM + n]  = v;
                proj_bf[row * DDIM + n] = f2bf(v);
            }
        }
    }
}

/* ---------- GEMM 2: h1 = relu(projected @ Wimp1^T + b1), N=128 ---------- */
__global__ __launch_bounds__(128)
void gemm_h1_kernel(const unsigned short* __restrict__ Pb,     /* [B, DDIM] */
                    const unsigned short* __restrict__ W1,     /* [128, DDIM] */
                    const float* __restrict__ bias,
                    float* __restrict__ h1) {                  /* [B, 128] */
    int lane = threadIdx.x & 31;
    int wave = threadIdx.x >> 5;
    int m0 = (blockIdx.x * 8 + wave * 2) * 16;
    int n0 = blockIdx.y * 64;

    v8f acc[2][4] = {};
    for (int k0 = 0; k0 < DDIM; k0 += 32) {
        v16bf a0 = load_A(Pb, DDIM, m0,      k0, lane);
        v16bf a1 = load_A(Pb, DDIM, m0 + 16, k0, lane);
        v16bf b0 = load_B(W1, DDIM, n0,      k0, lane);
        v16bf b1 = load_B(W1, DDIM, n0 + 16, k0, lane);
        v16bf b2 = load_B(W1, DDIM, n0 + 32, k0, lane);
        v16bf b3 = load_B(W1, DDIM, n0 + 48, k0, lane);
        acc[0][0] = WMMA_BF16(a0, b0, acc[0][0]);
        acc[0][1] = WMMA_BF16(a0, b1, acc[0][1]);
        acc[0][2] = WMMA_BF16(a0, b2, acc[0][2]);
        acc[0][3] = WMMA_BF16(a0, b3, acc[0][3]);
        acc[1][0] = WMMA_BF16(a1, b0, acc[1][0]);
        acc[1][1] = WMMA_BF16(a1, b1, acc[1][1]);
        acc[1][2] = WMMA_BF16(a1, b2, acc[1][2]);
        acc[1][3] = WMMA_BF16(a1, b3, acc[1][3]);
    }
    int col = lane & 15;
#pragma unroll
    for (int mi = 0; mi < 2; mi++) {
        int rbase = m0 + mi * 16 + ((lane & 16) ? 8 : 0);
#pragma unroll
        for (int t = 0; t < 4; t++) {
            int n = n0 + t * 16 + col;
            float bv = bias[n];
#pragma unroll
            for (int r = 0; r < 8; r++) {
                float v = acc[mi][t][r] + bv;
                h1[(rbase + r) * 128 + n] = v > 0.0f ? v : 0.0f;
            }
        }
    }
}

/* ---------- importance = sigmoid(h1 . w2 + b2) ---------- */
__global__ void importance_kernel(const float* __restrict__ h1, const float* __restrict__ w2,
                                  const float* __restrict__ b2, float* __restrict__ imp) {
    int b = blockIdx.x * blockDim.x + threadIdx.x;
    if (b >= B_ROWS) return;
    float s = 0.0f;
#pragma unroll 8
    for (int k = 0; k < 128; k++) s += h1[b * 128 + k] * w2[k];
    imp[b] = sigmoidf(s + b2[0]);
}

/* ---------- GEMM 3: fused GRU + gated blend (pre-LayerNorm) ----------
   grid (B/128, DDIM/16), block 128; wave = 32 rows x 16 cols, 12 accumulators.
   The W_ih gate tiles for this block's n0 (3 x 16 rows x 512 K = 48KB) are staged
   into LDS by the Tensor Data Mover (tensor_load_to_lds, TENSORcnt) when available. */
__global__ __launch_bounds__(128)
void gemm_gru_kernel(const unsigned short* __restrict__ Pb,    /* proj bf16 [B,512] */
                     const unsigned short* __restrict__ Tb,    /* team bf16 [B,512] */
                     const unsigned short* __restrict__ Wih,   /* [1536,512] bf16 */
                     const unsigned short* __restrict__ Whh,   /* [1536,512] bf16 */
                     const float* __restrict__ b_ih, const float* __restrict__ b_hh,
                     const float* __restrict__ team_f32,       /* [B,512] */
                     const float* __restrict__ imp,            /* [B] */
                     float* __restrict__ out_pre) {            /* [B,512] pre-LN */
    __shared__ unsigned short shB[3 * 16 * DDIM];              /* 48 KB */
    int lane = threadIdx.x & 31;
    int wave = threadIdx.x >> 5;
    int m0 = (blockIdx.x * 8 + wave * 2) * 16;
    int n0 = blockIdx.y * 16;

#if __has_builtin(__builtin_amdgcn_tensor_load_to_lds) && defined(__gfx1250__)
    if (wave == 0) {
        unsigned lbase = (unsigned)(unsigned long long)(const void*)&shB[0];
        for (int g = 0; g < 3; ++g) {
            unsigned long long ga =
                (unsigned long long)(const void*)(Wih + (size_t)(g * DDIM + n0) * DDIM);
            v4u g0;
            g0[0] = 1u;                                   /* count=1, user descriptor */
            g0[1] = lbase + (unsigned)(g * 16 * DDIM * 2);/* lds_addr (bytes) */
            g0[2] = (unsigned)(ga & 0xFFFFFFFFull);       /* global_addr[31:0] */
            g0[3] = (unsigned)((ga >> 32) & 0x1FFFFFFull) | (2u << 30); /* [56:32] | type=2 */
            v8i g1;
            g1[0] = 1 << 16;                 /* wg_mask=0, data_size=1 (2 bytes) */
            g1[1] = (int)(512u << 16);       /* tensor_dim0[15:0]=512 @bits[63:48] */
            g1[2] = (int)(1536u << 16);      /* tdim0 hi=0; tensor_dim1[15:0]=1536 */
            g1[3] = (int)(512u << 16);       /* tdim1 hi=0; tile_dim0=512 */
            g1[4] = 16;                      /* tile_dim1=16, tile_dim2=0 */
            g1[5] = 512;                     /* tensor_dim0_stride[31:0]=512 */
            g1[6] = 0;                       /* stride hi, dim1_stride lo */
            g1[7] = 0;
            v4i z4 = {0, 0, 0, 0};
            v8i z8 = {0, 0, 0, 0, 0, 0, 0, 0};
            __builtin_amdgcn_tensor_load_to_lds(g0, g1, z4, z4, z8, 0);
        }
        __builtin_amdgcn_s_wait_tensorcnt(0);
    }
    __syncthreads();
#else
    for (int v = threadIdx.x; v < 3 * 16 * (DDIM / 8); v += 128) {
        int col8 = v % (DDIM / 8);
        int lr   = v / (DDIM / 8);
        int g = lr >> 4, r = lr & 15;
        ((uint4*)shB)[v] = *(const uint4*)(Wih + (size_t)(g * DDIM + n0 + r) * DDIM + col8 * 8);
    }
    __syncthreads();
#endif

    v8f a_r[2] = {}, a_z[2] = {}, a_n[2] = {}, h_r[2] = {}, h_z[2] = {}, h_n[2] = {};
    for (int k0 = 0; k0 < DDIM; k0 += 32) {
        v16bf ap0 = load_A(Pb, DDIM, m0,      k0, lane);
        v16bf ap1 = load_A(Pb, DDIM, m0 + 16, k0, lane);
        v16bf at0 = load_A(Tb, DDIM, m0,      k0, lane);
        v16bf at1 = load_A(Tb, DDIM, m0 + 16, k0, lane);
        v16bf br = load_B_sh(shB,  0, k0, lane);
        v16bf bz = load_B_sh(shB, 16, k0, lane);
        v16bf bn = load_B_sh(shB, 32, k0, lane);
        v16bf cr = load_B(Whh, DDIM, n0,            k0, lane);
        v16bf cz = load_B(Whh, DDIM, n0 + DDIM,     k0, lane);
        v16bf cn = load_B(Whh, DDIM, n0 + 2 * DDIM, k0, lane);
        a_r[0] = WMMA_BF16(ap0, br, a_r[0]);  a_r[1] = WMMA_BF16(ap1, br, a_r[1]);
        a_z[0] = WMMA_BF16(ap0, bz, a_z[0]);  a_z[1] = WMMA_BF16(ap1, bz, a_z[1]);
        a_n[0] = WMMA_BF16(ap0, bn, a_n[0]);  a_n[1] = WMMA_BF16(ap1, bn, a_n[1]);
        h_r[0] = WMMA_BF16(at0, cr, h_r[0]);  h_r[1] = WMMA_BF16(at1, cr, h_r[1]);
        h_z[0] = WMMA_BF16(at0, cz, h_z[0]);  h_z[1] = WMMA_BF16(at1, cz, h_z[1]);
        h_n[0] = WMMA_BF16(at0, cn, h_n[0]);  h_n[1] = WMMA_BF16(at1, cn, h_n[1]);
    }

    int col = n0 + (lane & 15);
    float bir = b_ih[col], biz = b_ih[DDIM + col], bin = b_ih[2 * DDIM + col];
    float bhr = b_hh[col], bhz = b_hh[DDIM + col], bhn = b_hh[2 * DDIM + col];
#pragma unroll
    for (int mi = 0; mi < 2; mi++) {
        int rbase = m0 + mi * 16 + ((lane & 16) ? 8 : 0);
#pragma unroll
        for (int r = 0; r < 8; r++) {
            int row = rbase + r;
            float rg = sigmoidf((a_r[mi][r] + bir) + (h_r[mi][r] + bhr));
            float zg = sigmoidf((a_z[mi][r] + biz) + (h_z[mi][r] + bhz));
            float ng = tanhf((a_n[mi][r] + bin) + rg * (h_n[mi][r] + bhn));
            float te = team_f32[row * DDIM + col];
            float ne = (1.0f - zg) * ng + zg * te;
            float im = imp[row];
            out_pre[row * DDIM + col] = te + im * (ne - te);
        }
    }
}

/* ---------- in-place row LayerNorm over d_out[0 .. B*D) ---------- */
__global__ __launch_bounds__(256)
void layernorm_kernel(float* __restrict__ out, const float* __restrict__ gamma,
                      const float* __restrict__ beta) {
    int lane = threadIdx.x & 31;
    int wave = threadIdx.x >> 5;
    int row = blockIdx.x * 8 + wave;
    if (row >= B_ROWS) return;
    float* p = out + row * DDIM;
    float v[16];
    float s = 0.0f;
#pragma unroll
    for (int i = 0; i < 16; i++) { v[i] = p[lane + i * 32]; s += v[i]; }
#pragma unroll
    for (int off = 16; off > 0; off >>= 1) s += __shfl_xor(s, off, 32);
    float mu = s * (1.0f / (float)DDIM);
    float var = 0.0f;
#pragma unroll
    for (int i = 0; i < 16; i++) { float d = v[i] - mu; var += d * d; }
#pragma unroll
    for (int off = 16; off > 0; off >>= 1) var += __shfl_xor(var, off, 32);
    float inv = rsqrtf(var * (1.0f / (float)DDIM) + LN_EPS);
#pragma unroll
    for (int i = 0; i < 16; i++) {
        int c = lane + i * 32;
        p[c] = gamma[c] * (v[i] - mu) * inv + beta[c];
    }
}

/* ---------- launcher ---------- */

extern "C" void kernel_launch(void* const* d_in, const int* in_sizes, int n_in,
                              void* d_out, int out_size, void* d_ws, size_t ws_size,
                              hipStream_t stream) {
    const float* team = (const float*)d_in[0];
    const float* opp  = (const float*)d_in[1];
    const float* gf   = (const float*)d_in[2];
    const float* won  = (const float*)d_in[3];
    const float* Wp   = (const float*)d_in[4];
    const float* bp   = (const float*)d_in[5];
    const float* Wih  = (const float*)d_in[6];
    const float* bih  = (const float*)d_in[7];
    const float* Whh  = (const float*)d_in[8];
    const float* bhh  = (const float*)d_in[9];
    const float* gam  = (const float*)d_in[10];
    const float* bet  = (const float*)d_in[11];
    const float* W1   = (const float*)d_in[12];
    const float* b1   = (const float*)d_in[13];
    const float* W2   = (const float*)d_in[14];
    const float* b2   = (const float*)d_in[15];

    float* updated = (float*)d_out;                         /* [B, D] */
    float* gc_out  = (float*)d_out + (size_t)B_ROWS * DDIM; /* [B, D] = projected */

    /* workspace carve-up */
    char* ws = (char*)d_ws;
    unsigned short* Xb   = (unsigned short*)ws;  ws += (size_t)B_ROWS * KPAD * 2;
    unsigned short* Tb   = (unsigned short*)ws;  ws += (size_t)B_ROWS * DDIM * 2;
    unsigned short* Wpb  = (unsigned short*)ws;  ws += (size_t)DDIM * KPAD * 2;
    unsigned short* Wihb = (unsigned short*)ws;  ws += (size_t)G3 * DDIM * 2;
    unsigned short* Whhb = (unsigned short*)ws;  ws += (size_t)G3 * DDIM * 2;
    unsigned short* W1b  = (unsigned short*)ws;  ws += (size_t)128 * DDIM * 2;
    unsigned short* Pb   = (unsigned short*)ws;  ws += (size_t)B_ROWS * DDIM * 2;
    float*          h1   = (float*)ws;           ws += (size_t)B_ROWS * 128 * 4;
    float*          imp  = (float*)ws;           ws += (size_t)B_ROWS * 4;

    /* 1. pack + convert to bf16 */
    {
        int total = B_ROWS * KPAD;
        pack_x_kernel<<<(total + 255) / 256, 256, 0, stream>>>(team, opp, gf, won, Xb, total);
    }
    {
        int total = B_ROWS * DDIM;
        convert_pad_kernel<<<(total + 255) / 256, 256, 0, stream>>>(team, Tb, DDIM, DDIM, total);
    }
    {
        int total = DDIM * KPAD;
        convert_pad_kernel<<<(total + 255) / 256, 256, 0, stream>>>(Wp, Wpb, 1053, KPAD, total);
    }
    {
        int total = G3 * DDIM;
        convert_pad_kernel<<<(total + 255) / 256, 256, 0, stream>>>(Wih, Wihb, DDIM, DDIM, total);
        convert_pad_kernel<<<(total + 255) / 256, 256, 0, stream>>>(Whh, Whhb, DDIM, DDIM, total);
    }
    {
        int total = 128 * DDIM;
        convert_pad_kernel<<<(total + 255) / 256, 256, 0, stream>>>(W1, W1b, DDIM, DDIM, total);
    }

    /* 2. projection GEMM -> gc_out (fp32, = game_context output) + Pb (bf16) */
    gemm_proj_kernel<<<dim3(B_ROWS / 128, DDIM / 64), 128, 0, stream>>>(Xb, Wpb, bp, gc_out, Pb);

    /* 3. importance MLP */
    gemm_h1_kernel<<<dim3(B_ROWS / 128, 128 / 64), 128, 0, stream>>>(Pb, W1b, b1, h1);
    importance_kernel<<<(B_ROWS + 255) / 256, 256, 0, stream>>>(h1, W2, b2, imp);

    /* 4. fused GRU GEMM + gated blend -> updated (pre-LN) */
    gemm_gru_kernel<<<dim3(B_ROWS / 128, DDIM / 16), 128, 0, stream>>>(
        Pb, Tb, Wihb, Whhb, bih, bhh, team, imp, updated);

    /* 5. in-place LayerNorm */
    layernorm_kernel<<<B_ROWS / 8, 256, 0, stream>>>(updated, gam, bet);

    (void)in_sizes; (void)n_in; (void)out_size; (void)ws_size;
}
